// StratifiedRecurrence_63960652972054
// MI455X (gfx1250) — matrix operand order repaired and checked by reading, not compiled, not run on hardware
//
#include <hip/hip_runtime.h>
#include <hip/hip_bf16.h>

#define C_   8
#define B_   4
#define T_   4096
#define D_   128
#define SEGS 16
#define SEGL (T_ / SEGS)   // 256

typedef __attribute__((ext_vector_type(16))) __bf16 v16bf;
typedef __attribute__((ext_vector_type(8)))  float  v8f;

__device__ __forceinline__ float lam_of(int c) {
    // 1 - 2^(-c * log2(8192)/7),  c=0 -> 0
    return 1.0f - exp2f(-(float)c * (13.0f / 7.0f));
}

// ---------------------------------------------------------------------------
// Kernel 0: W (C,D,D) f32  ->  Wt (2,C,D,D) bf16, N-major:  wt[dir][c][e][d]
// ---------------------------------------------------------------------------
__global__ void wt_convert_kernel(const float* __restrict__ Wf,
                                  const float* __restrict__ Wb,
                                  __bf16* __restrict__ wt) {
    int idx = blockIdx.x * blockDim.x + threadIdx.x;
    const int total = 2 * C_ * D_ * D_;
    if (idx >= total) return;
    int dk = idx % D_;
    int t  = idx / D_;
    int e  = t % D_;  t /= D_;
    int c  = t % C_;
    int dir = t / C_;
    const float* W = dir ? Wb : Wf;
    wt[idx] = (__bf16)(W[((size_t)c * D_ + dk) * D_ + e]);
}

// ---------------------------------------------------------------------------
// Kernel 1: per-wave 16x128 tile of  v = sigmoid(z @ W + b) * z
//           accumulation via v_wmma_f32_16x16x32_bf16 (4 K-steps x 8 N-tiles)
// ---------------------------------------------------------------------------
__global__ void __launch_bounds__(128)
gemm_gate_kernel(const float* __restrict__ z_all,     // (C,B,T,D) f32
                 const __bf16* __restrict__ wt_all,   // (2,C,D,D) bf16 N-major
                 const float* __restrict__ bf,
                 const float* __restrict__ bb,
                 float* __restrict__ out)             // (2,C,B,T,D) f32
{
    const int wave = threadIdx.x >> 5;
    const int lane = threadIdx.x & 31;
    const int half = lane >> 4;      // 0: lanes 0-15, 1: lanes 16-31
    const int lm   = lane & 15;
    const int c    = blockIdx.y;
    const int dir  = blockIdx.z;
    const int mtile = blockIdx.x * 4 + wave;   // 0..1023 (16-row tiles of B*T)
    const int row0  = mtile * 16;

    const float*  zc   = z_all  + (size_t)c * ((size_t)B_ * T_ * D_);
    const __bf16* wt   = wt_all + ((size_t)dir * C_ + c) * ((size_t)D_ * D_);
    const float*  bias = (dir ? bb : bf) + c * D_;
    float*        oc   = out + ((size_t)dir * C_ + c) * ((size_t)B_ * T_ * D_);

    v8f acc[8] = {};   // 8 N-tiles of 16x16 f32

    // A operand: row M = row0+lm for this lane (both halves), per ISA layout.
    const float* arow = zc + (size_t)(row0 + lm) * D_;

    #pragma unroll
    for (int ks = 0; ks < 4; ++ks) {             // K = 32*ks .. 32*ks+31
        v16bf A;
        const int ka = ks * 32 + half * 8;
        #pragma unroll
        for (int j = 0; j < 8; ++j) A[j]     = (__bf16)arow[ka + j];        // K = ka..ka+7
        #pragma unroll
        for (int j = 0; j < 8; ++j) A[8 + j] = (__bf16)arow[ka + 16 + j];   // K = ka+16..

        #pragma unroll
        for (int nt = 0; nt < 8; ++nt) {         // N-tile
            // B operand: lane holds N = nt*16+lm, K = ks*32 + half*8 + (0..7), +16 + (0..7)
            union { v16bf v; uint4 q[2]; } Bm;
            const __bf16* wrow =
                wt + (size_t)(nt * 16 + lm) * D_ + (ks * 32 + half * 8);
            Bm.q[0] = *(const uint4*)(wrow);
            Bm.q[1] = *(const uint4*)(wrow + 16);
            acc[nt] = __builtin_amdgcn_wmma_f32_16x16x32_bf16(
                false, A, false, Bm.v, (short)0, acc[nt], false, false);
        }
    }

    // Epilogue: C/D layout -> lane(half,lm), vgpr r: M = row0 + r + half*8, N = nt*16+lm
    #pragma unroll
    for (int nt = 0; nt < 8; ++nt) {
        const int   N  = nt * 16 + lm;
        const float bv = bias[N];
        #pragma unroll
        for (int r = 0; r < 8; ++r) {
            const int M = row0 + r + half * 8;
            float x = acc[nt][r] + bv;
            float s = __builtin_amdgcn_rcpf(1.0f + exp2f(x * -1.44269504f));
            float zv = zc[(size_t)M * D_ + N];
            oc[(size_t)M * D_ + N] = s * zv;
        }
    }
}

// ---------------------------------------------------------------------------
// Kernel 2: segment-local scans  h_j = lam*h_{j-1} + v_j  (in place in d_out),
//           segment-final h -> ends[dir][c][b][seg][d]
// ---------------------------------------------------------------------------
__global__ void __launch_bounds__(128)
scan_local_kernel(float* __restrict__ out, float* __restrict__ ends) {
    int blk = blockIdx.x;
    const int seg = blk % SEGS; blk /= SEGS;
    const int b   = blk % B_;   blk /= B_;
    const int c   = blk % C_;
    const int dir = blk / C_;
    const int d   = threadIdx.x;
    const float lam = lam_of(c);

    float* p = out + ((size_t)dir * C_ + c) * ((size_t)B_ * T_ * D_)
                   + (size_t)b * T_ * D_ + d;
    const long st = dir ? -(long)D_ : (long)D_;
    const long t0 = dir ? (long)(T_ - 1 - seg * SEGL) : (long)(seg * SEGL);
    float* q = p + t0 * (long)D_;

    float h = 0.0f;
    for (int j = 0; j < SEGL; j += 16) {
        float vbuf[16];
        #pragma unroll
        for (int k = 0; k < 16; ++k) vbuf[k] = q[(long)(j + k) * st];
        #pragma unroll
        for (int k = 0; k < 16; ++k) {
            h = fmaf(lam, h, vbuf[k]);
            q[(long)(j + k) * st] = h;
        }
    }
    ends[((((size_t)dir * C_ + c) * B_ + b) * SEGS + seg) * D_ + d] = h;
}

// ---------------------------------------------------------------------------
// Kernel 3: cross-segment fix-up:  out[j] += lam^(j+1) * H_prev
//           H_prev rebuilt per block from upstream segment ends.
// ---------------------------------------------------------------------------
__global__ void __launch_bounds__(128)
scan_fix_kernel(float* __restrict__ out, const float* __restrict__ ends) {
    int blk = blockIdx.x;
    const int seg = blk % SEGS; blk /= SEGS;
    const int b   = blk % B_;   blk /= B_;
    const int c   = blk % C_;
    const int dir = blk / C_;
    if (seg == 0) return;
    const int d = threadIdx.x;
    const float lam = lam_of(c);

    float lamL = 1.0f;
    #pragma unroll 8
    for (int i = 0; i < SEGL; ++i) lamL *= lam;   // lam^SEGL

    const float* e = ends + ((((size_t)dir * C_ + c) * B_ + b) * SEGS) * D_ + d;
    float Hp = 0.0f;
    for (int s = 0; s < seg; ++s) Hp = fmaf(Hp, lamL, e[(size_t)s * D_]);

    float* p = out + ((size_t)dir * C_ + c) * ((size_t)B_ * T_ * D_)
                   + (size_t)b * T_ * D_ + d;
    const long st = dir ? -(long)D_ : (long)D_;
    const long t0 = dir ? (long)(T_ - 1 - seg * SEGL) : (long)(seg * SEGL);
    float* q = p + t0 * (long)D_;

    float f = lam;
    for (int j = 0; j < SEGL; j += 8) {
        float fb[8], vb[8];
        #pragma unroll
        for (int k = 0; k < 8; ++k) { fb[k] = f; f *= lam; }
        #pragma unroll
        for (int k = 0; k < 8; ++k) vb[k] = q[(long)(j + k) * st];
        #pragma unroll
        for (int k = 0; k < 8; ++k) q[(long)(j + k) * st] = fmaf(fb[k], Hp, vb[k]);
    }
}

// ---------------------------------------------------------------------------
extern "C" void kernel_launch(void* const* d_in, const int* in_sizes, int n_in,
                              void* d_out, int out_size, void* d_ws, size_t ws_size,
                              hipStream_t stream) {
    const float* channels = (const float*)d_in[0];
    const float* Wf = (const float*)d_in[1];
    const float* bf = (const float*)d_in[2];
    const float* Wb = (const float*)d_in[3];
    const float* bb = (const float*)d_in[4];
    float* out = (float*)d_out;

    float* ends = (float*)d_ws;  // 2*C*B*SEGS*D floats = 512 KB
    __bf16* wt = (__bf16*)((char*)d_ws +
                 (size_t)2 * C_ * B_ * SEGS * D_ * sizeof(float));

    {   // 0) W -> bf16, N-major
        int total = 2 * C_ * D_ * D_;
        wt_convert_kernel<<<(total + 255) / 256, 256, 0, stream>>>(Wf, Wb, wt);
    }
    {   // 1) WMMA gemm + gate, writes v into d_out
        dim3 grid(B_ * T_ / 16 / 4, C_, 2);   // (256, 8, 2), 4 waves/block
        gemm_gate_kernel<<<grid, 128, 0, stream>>>(channels, wt, bf, bb, out);
    }
    // 2) segment-local scans (in place) + carries
    scan_local_kernel<<<2 * C_ * B_ * SEGS, 128, 0, stream>>>(out, ends);
    // 3) cross-segment fix-up
    scan_fix_kernel<<<2 * C_ * B_ * SEGS, 128, 0, stream>>>(out, ends);
}
